// QLSTMHarmonic_23304492548464
// MI455X (gfx1250) — matrix operand order repaired and checked by reading, not compiled, not run on hardware
//
#include <hip/hip_runtime.h>
#include <hip/hip_bf16.h>

typedef __attribute__((ext_vector_type(16))) _Float16 v16h;
typedef __attribute__((ext_vector_type(8)))  _Float16 v8h;
typedef __attribute__((ext_vector_type(8)))  float    v8f;

#define Q_SEQ 64
#define Q_HID 20
#define Q_NG  80   // 4*HID

// ---------------- fast transcendental helpers (guarded; gfx1250 has native TRANS ops) ----
__device__ __forceinline__ float fast_rcp(float x) {
#if __has_builtin(__builtin_amdgcn_rcpf)
    return __builtin_amdgcn_rcpf(x);          // v_rcp_f32
#else
    return 1.0f / x;
#endif
}
__device__ __forceinline__ float fast_exp2(float x) {
#if __has_builtin(__builtin_amdgcn_exp2f)
    return __builtin_amdgcn_exp2f(x);         // v_exp_f32 (2^x)
#else
    return exp2f(x);
#endif
}
__device__ __forceinline__ float fast_tanh(float x) {
#if __has_builtin(__builtin_amdgcn_tanhf)
    return __builtin_amdgcn_tanhf(x);         // v_tanh_f32 (gfx1250 TRANS)
#else
    return tanhf(x);
#endif
}

// ---------------- reduction helpers ----------------
__device__ __forceinline__ float warp_max(float v) {
#pragma unroll
    for (int off = 16; off > 0; off >>= 1)
        v = fmaxf(v, __shfl_xor(v, off, 32));
    return v;
}

// block-wide max over 256 threads (8 waves). All threads must call.
__device__ __forceinline__ float block_max(float v, float* sred) {
    v = warp_max(v);
    if ((threadIdx.x & 31) == 0) sred[threadIdx.x >> 5] = v;
    __syncthreads();
    float r = fmaxf(fmaxf(fmaxf(sred[0], sred[1]), fmaxf(sred[2], sred[3])),
                    fmaxf(fmaxf(sred[4], sred[5]), fmaxf(sred[6], sred[7])));
    __syncthreads();
    return r;
}

// ---------------- fake-quant helpers (scale + precomputed reciprocal) ----------------
__device__ __forceinline__ float fq_i(float x, float s, float inv_s, float qmax) {
    float r = rintf(x * inv_s);
    r = fminf(fmaxf(r, -qmax), qmax);
    return r * s;
}
__device__ __forceinline__ float fq_u(float x, float s, float inv_s, float qmax) {
    // x >= 0 (post-ReLU)
    return fminf(rintf(x * inv_s), qmax) * s;
}
__device__ __forceinline__ float qsig6(float x) {       // quantized sigmoid, 6-bit
    float y = fast_rcp(1.0f + fast_exp2(x * -1.4426950408889634f));
    return rintf(y * 63.0f) * (1.0f / 63.0f);
}
__device__ __forceinline__ float qtanh6(float x) {      // quantized tanh, 6-bit
    float y = fast_tanh(x);
    return rintf(y * 31.0f) * (1.0f / 31.0f);
}

// ---------------- pre-pass: global max|x| ----------------
__global__ void ws_zero_kernel(unsigned int* ws) {
    if (threadIdx.x == 0) ws[0] = 0u;
}

__global__ __launch_bounds__(256) void xmax_kernel(const float* __restrict__ x, int n,
                                                   unsigned int* __restrict__ ws) {
    __shared__ float sred[8];
    float v = 0.0f;
    for (int i = blockIdx.x * 256 + threadIdx.x; i < n; i += gridDim.x * 256)
        v = fmaxf(v, fabsf(x[i]));
    v = warp_max(v);
    if ((threadIdx.x & 31) == 0) sred[threadIdx.x >> 5] = v;
    __syncthreads();
    if (threadIdx.x == 0) {
        float r = sred[0];
#pragma unroll
        for (int i = 1; i < 8; i++) r = fmaxf(r, sred[i]);
        atomicMax(ws, __float_as_uint(r));   // valid ordering: r >= 0
    }
}

// ---------------- main fused QLSTM + MLP kernel ----------------
// 256 threads = 8 waves; each wave owns 16 batch rows (block: 128 rows).
__global__ __launch_bounds__(256) void qlstm_kernel(
    const float* __restrict__ x,    const float* __restrict__ w_ih,
    const float* __restrict__ w_hh, const float* __restrict__ bias,
    const float* __restrict__ W1,   const float* __restrict__ b1,
    const float* __restrict__ W2,   const float* __restrict__ b2,
    const float* __restrict__ W3,   const float* __restrict__ b3,
    float* __restrict__ out,        const unsigned int* __restrict__ ws)
{
    __shared__ _Float16 hstage[8 * 16 * 32];   // per-wave h staging, K padded to 32 (8 KB)
    __shared__ float    gsh[8 * 16 * 80];      // per-wave raw gates (40 KB); reused by MLP
    __shared__ float    w1q[64 * 20];
    __shared__ float    w2q[32 * 64];
    __shared__ float    w3q[4 * 32];
    __shared__ float    sred[8];

    const int tid  = threadIdx.x;
    const int wv   = tid >> 5;        // wave id 0..7
    const int lane = tid & 31;
    const int half = lane >> 4;       // 0: lanes 0-15, 1: lanes 16-31
    const int col  = lane & 15;       // N column / A row within tile
    const int rowBase = blockIdx.x * 128 + wv * 16;

    // ---- per-tensor weight scales (exact: every block reduces the full tensors)
    float m_ih = 0.f, m_hh = 0.f, mW1 = 0.f, mW2 = 0.f, mW3 = 0.f;
    for (int i = tid; i < Q_NG;         i += 256) m_ih = fmaxf(m_ih, fabsf(w_ih[i]));
    for (int i = tid; i < Q_NG * Q_HID; i += 256) m_hh = fmaxf(m_hh, fabsf(w_hh[i]));
    for (int i = tid; i < 64 * 20;      i += 256) mW1  = fmaxf(mW1, fabsf(W1[i]));
    for (int i = tid; i < 32 * 64;      i += 256) mW2  = fmaxf(mW2, fabsf(W2[i]));
    for (int i = tid; i < 4 * 32;       i += 256) mW3  = fmaxf(mW3, fabsf(W3[i]));
    const float M_ih = fmaxf(block_max(m_ih, sred), 1e-8f);
    const float M_hh = fmaxf(block_max(m_hh, sred), 1e-8f);
    const float MW1  = fmaxf(block_max(mW1, sred), 1e-8f);
    const float MW2  = fmaxf(block_max(mW2, sred), 1e-8f);
    const float MW3  = fmaxf(block_max(mW3, sred), 1e-8f);
    const float s_ih = M_ih / 127.0f, i_ih = 127.0f * fast_rcp(M_ih);
    const float s_hh = M_hh / 127.0f, i_hh = 127.0f * fast_rcp(M_hh);
    const float sW1  = MW1 / 127.0f,  iW1  = 127.0f * fast_rcp(MW1);
    const float sW2  = MW2 / 127.0f,  iW2  = 127.0f * fast_rcp(MW2);
    const float sW3  = MW3 / 127.0f,  iW3  = 127.0f * fast_rcp(MW3);
    const float Mx   = fmaxf(__uint_as_float(ws[0]), 1e-8f);     // global |x| max
    const float sx   = Mx / 127.0f,   ix   = 127.0f * fast_rcp(Mx);

    // ---- pre-quantize MLP weights into LDS
    for (int i = tid; i < 64 * 20; i += 256) w1q[i] = fq_i(W1[i], sW1, iW1, 127.0f);
    for (int i = tid; i < 32 * 64; i += 256) w2q[i] = fq_i(W2[i], sW2, iW2, 127.0f);
    for (int i = tid; i < 4 * 32;  i += 256) w3q[i] = fq_i(W3[i], sW3, iW3, 127.0f);

    // ---- build resident B fragments: B[k][n] = w_hh_q[n][k], row 20 = w_ih_q[n]
    //      layout (16-bit B 32x16): lane col = n, halfs q -> K = 8*half + q (q<8), 16 + 8*half + (q-8) (q>=8)
    v16h  bfrag[5];
    float bias5[5];
#pragma unroll
    for (int t = 0; t < 5; t++) {
        const int n = t * 16 + col;
        bias5[t] = bias[n];
#pragma unroll
        for (int q = 0; q < 16; q++) {
            const int k = (q < 8) ? (8 * half + q) : (16 + 8 * half + (q - 8));
            float v;
            if (k < Q_HID)       v = fq_i(w_hh[n * Q_HID + k], s_hh, i_hh, 127.0f);
            else if (k == Q_HID) v = fq_i(w_ih[n], s_ih, i_ih, 127.0f);
            else                 v = 0.0f;
            bfrag[t][q] = (_Float16)v;
        }
    }

    // ---- init h staging: zeros, x[.,0] (io-quantized) in K column 20
#pragma unroll
    for (int q = 0; q < 16; q++)
        hstage[wv * 512 + lane * 16 + q] = (_Float16)0.0f;
    __syncthreads();
    if (lane < 16) {
        const float xv = x[(size_t)(rowBase + lane) * Q_SEQ];
        hstage[wv * 512 + lane * 32 + Q_HID] = (_Float16)fq_i(xv, sx, ix, 127.0f);
    }
    __syncthreads();

    // per-lane cell state: linear index p = lane*10+q -> (m = p/20, j = p%20)
    float c_reg[10];
#pragma unroll
    for (int q = 0; q < 10; q++) c_reg[q] = 0.0f;

    // ================= recurrent loop =================
    for (int t = 0; t < Q_SEQ; t++) {
        // A fragment: lane holds row m=col, K chunks [8h, 8h+8) and [8h+16, 8h+24)
        const v8h* hp = (const v8h*)(hstage + wv * 512 + col * 32 + half * 8);
        union { v16h v; v8h h[2]; } au;
        au.h[0] = hp[0];
        au.h[1] = hp[2];    // +16 halfs
        const v16h afrag = au.v;

        float gm = 0.0f;
#pragma unroll
        for (int tt = 0; tt < 5; tt++) {
            v8f ci;
#pragma unroll
            for (int k = 0; k < 8; k++) ci[k] = bias5[tt];
            v8f acc = __builtin_amdgcn_wmma_f32_16x16x32_f16(
                false, afrag, false, bfrag[tt], (short)0, ci, false, false);
#pragma unroll
            for (int v = 0; v < 8; v++) {
                const float g = acc[v];
                const int m = v + 8 * half;
                gsh[wv * 1280 + m * 80 + tt * 16 + col] = g;
                gm = fmaxf(gm, fabsf(g));
            }
        }
        // dynamic 6-bit gate scale (per-block approximation of batch-global scale)
        const float Mg = fmaxf(block_max(gm, sred), 1e-8f);
        const float sg = Mg * (1.0f / 31.0f);
        const float ig = 31.0f * fast_rcp(Mg);

        float cn[10], ov[10];
        float cm = 0.0f;
#pragma unroll
        for (int q = 0; q < 10; q++) {
            const int p = lane * 10 + q, m = p / 20, j = p % 20;
            const float* gr = gsh + wv * 1280 + m * 80 + j;
            const float gi = fq_i(gr[0],  sg, ig, 31.0f);
            const float gf = fq_i(gr[20], sg, ig, 31.0f);
            const float gg = fq_i(gr[40], sg, ig, 31.0f);
            const float go = fq_i(gr[60], sg, ig, 31.0f);
            const float iv = qsig6(gi), fv = qsig6(gf), gv = qtanh6(gg);
            ov[q] = qsig6(go);
            const float c2 = fv * c_reg[q] + iv * gv;
            cn[q] = c2;
            cm = fmaxf(cm, fabsf(c2));
        }
        // dynamic 6-bit cell-state scale
        const float Mc = fmaxf(block_max(cm, sred), 1e-8f);
        const float sc = Mc * (1.0f / 31.0f);
        const float ic = 31.0f * fast_rcp(Mc);
#pragma unroll
        for (int q = 0; q < 10; q++) {
            const int p = lane * 10 + q, m = p / 20, j = p % 20;
            const float c2 = fq_i(cn[q], sc, ic, 31.0f);
            c_reg[q] = c2;
            const float hv = ov[q] * qtanh6(c2);
            hstage[wv * 512 + m * 32 + j] = (_Float16)hv;
            if (j == 0 && t + 1 < Q_SEQ) {
                const float xv = x[(size_t)(rowBase + m) * Q_SEQ + (t + 1)];
                hstage[wv * 512 + m * 32 + Q_HID] = (_Float16)fq_i(xv, sx, ix, 127.0f);
            }
        }
        __syncthreads();
    }

    // ================= MLP head: one row per thread (rows 0..127) =================
    const bool active = tid < 128;
    float a0[20];
    float am = 0.0f;
    if (active) {
        const _Float16* hp = hstage + (tid >> 4) * 512 + (tid & 15) * 32;
#pragma unroll
        for (int j = 0; j < Q_HID; j++) {
            a0[j] = fmaxf((float)hp[j], 0.0f);
            am = fmaxf(am, a0[j]);
        }
    }
    const float M0 = fmaxf(block_max(am, sred), 1e-8f);
    const float s0 = M0 * (1.0f / 63.0f), i0 = 63.0f * fast_rcp(M0);

    float om1 = 0.0f;
    if (active) {
#pragma unroll
        for (int j = 0; j < Q_HID; j++)
            a0[j] = fq_u(a0[j], s0, i0, 63.0f);
        for (int jo = 0; jo < 64; jo++) {
            float acc = b1[jo];
#pragma unroll
            for (int k = 0; k < Q_HID; k++) acc += a0[k] * w1q[jo * 20 + k];
            const float r = fmaxf(acc, 0.0f);
            gsh[tid * 64 + jo] = r;          // reuse gates buffer
            om1 = fmaxf(om1, r);
        }
    }
    const float M1 = fmaxf(block_max(om1, sred), 1e-8f);
    const float s1 = M1 * (1.0f / 63.0f), i1 = 63.0f * fast_rcp(M1);

    float o2[32];
    float om2 = 0.0f;
    if (active) {
        for (int k = 0; k < 64; k++)
            gsh[tid * 64 + k] = fq_u(gsh[tid * 64 + k], s1, i1, 63.0f);
        for (int jo = 0; jo < 32; jo++) {
            float acc = b2[jo];
            for (int k = 0; k < 64; k++) acc += gsh[tid * 64 + k] * w2q[jo * 64 + k];
            o2[jo] = fmaxf(acc, 0.0f);
            om2 = fmaxf(om2, o2[jo]);
        }
    }
    const float M2 = fmaxf(block_max(om2, sred), 1e-8f);
    const float s2 = M2 * (1.0f / 63.0f), i2 = 63.0f * fast_rcp(M2);

    if (active) {
#pragma unroll
        for (int k = 0; k < 32; k++)
            o2[k] = fq_u(o2[k], s2, i2, 63.0f);
        const size_t orow = (size_t)(blockIdx.x * 128 + tid) * 4;
#pragma unroll
        for (int jo = 0; jo < 4; jo++) {
            float acc = b3[jo];
#pragma unroll
            for (int k = 0; k < 32; k++) acc += o2[k] * w3q[jo * 32 + k];
            out[orow + jo] = fmaxf(acc, 0.0f);
        }
    }
}

// ---------------- host launch ----------------
extern "C" void kernel_launch(void* const* d_in, const int* in_sizes, int n_in,
                              void* d_out, int out_size, void* d_ws, size_t ws_size,
                              hipStream_t stream) {
    (void)n_in; (void)out_size; (void)ws_size;
    const float* x    = (const float*)d_in[0];
    const float* w_ih = (const float*)d_in[1];
    const float* w_hh = (const float*)d_in[2];
    const float* b    = (const float*)d_in[3];
    const float* W1   = (const float*)d_in[4];
    const float* b1   = (const float*)d_in[5];
    const float* W2   = (const float*)d_in[6];
    const float* b2   = (const float*)d_in[7];
    const float* W3   = (const float*)d_in[8];
    const float* b3   = (const float*)d_in[9];
    unsigned int* ws  = (unsigned int*)d_ws;
    float* out        = (float*)d_out;

    const int nX = in_sizes[0];          // B * SEQ
    const int B  = nX / Q_SEQ;           // 131072

    ws_zero_kernel<<<1, 64, 0, stream>>>(ws);
    xmax_kernel<<<512, 256, 0, stream>>>(x, nX, ws);
    qlstm_kernel<<<B / 128, 256, 0, stream>>>(x, w_ih, w_hh, b,
                                              W1, b1, W2, b2, W3, b3, out, ws);
}
